// AWDLSTM_11038065951295
// MI455X (gfx1250) — compile-verified, matrix operand order checked by reading, not run on hardware
//
#include <hip/hip_runtime.h>
#include <hip/hip_bf16.h>
#include <stdint.h>

// ---------------- problem constants ----------------
#define Hh   1150          // hidden
#define HP   1152          // hidden padded to 16/32 multiple
#define Ee   400           // embedding dim (mult of 16)
#define EP   416           // embedding dim padded to mult of 32
#define Bb   32            // batch
#define Ss   512           // sequence
#define Vv   10000         // vocab

typedef __bf16 bf16_t;
typedef __attribute__((ext_vector_type(16))) __bf16 v16bf;
typedef __attribute__((ext_vector_type(8)))  __bf16 v8bf;
typedef __attribute__((ext_vector_type(8)))  float   v8f;

// float -> bf16 round-to-nearest-even via bit manipulation (compile-safe)
__device__ __forceinline__ bf16_t f2bf(float f) {
  union { float f; uint32_t u; } v; v.f = f;
  uint32_t r = v.u + 0x7FFFu + ((v.u >> 16) & 1u);
  uint16_t h = (uint16_t)(r >> 16);
  bf16_t o; __builtin_memcpy(&o, &h, 2);
  return o;
}

// Load one lane's share of a 16x32 bf16 WMMA fragment from row-major memory.
// p must already point at element [row][k0 + (lane>=16 ? 8 : 0)].
// Lane holds K {0..7,16..23} (lanes 0-15) or {8..15,24..31} (lanes 16-31),
// i.e. two contiguous 16-byte runs.  Works for global or LDS pointers.
__device__ __forceinline__ v16bf load_frag(const bf16_t* p) {
  v8bf lo = *(const v8bf*)(p);
  v8bf hi = *(const v8bf*)(p + 16);
  return __builtin_shufflevector(lo, hi, 0,1,2,3,4,5,6,7,8,9,10,11,12,13,14,15);
}

__device__ __forceinline__ v8f wmma_bf16(v16bf a, v16bf b, v8f c) {
  // D(f32) = A(bf16 16x32) * B(bf16 32x16) + C(f32)
  return __builtin_amdgcn_wmma_f32_16x16x32_bf16(false, a, false, b, (short)0, c,
                                                 false, false);
}

__device__ __forceinline__ float sigm(float x) { return 1.0f / (1.0f + __expf(-x)); }

// Issue one lane's 16-byte async global->LDS copy (ASYNCcnt-tracked DMA).
__device__ __forceinline__ void async_g2l_b128(uint32_t lds_off, const void* gaddr) {
  asm volatile("global_load_async_to_lds_b128 %0, %1, off"
               :: "v"(lds_off), "v"((uint64_t)(uintptr_t)gaddr)
               : "memory");
}
__device__ __forceinline__ void wait_async0() {
  asm volatile("s_wait_asynccnt 0x0" ::: "memory");
}

// =====================================================================
// Fused LSTM step: gates = x@Wih^T + h@Whh^T + b_ih + b_hh, then cell
// update.  One wave computes a 16(M)x16(N) tile of ALL FOUR gates
// (weights pre-packed per-gate, zero padded to [4][HP][Kp]), then does
// the sigmoid/tanh cell update in-register.  144 waves total (2 M-tiles
// x 72 N-tiles).
// =====================================================================
__global__ void __launch_bounds__(256) lstm_step_kernel(
    const bf16_t* __restrict__ xin, int ldx, int Kx,   // A input [32][ldx]
    const bf16_t* __restrict__ wih,                    // [4][HP][Kx]
    const bf16_t* __restrict__ hprev,                  // [32][HP]
    const bf16_t* __restrict__ whh,                    // [4][HP][HP]
    const float*  __restrict__ bih,                    // [4*Hh]
    const float*  __restrict__ bhh,                    // [4*Hh]
    float*  __restrict__ c_state,                      // [32][HP]
    float*  __restrict__ h_state,                      // [32][HP]
    bf16_t* __restrict__ hnew,                         // [32][HP] (ping-pong)
    float*  __restrict__ raw_f32, long raw_f32_stride, // layer2 only (else null)
    bf16_t* __restrict__ raw_bf,  long raw_bf_stride)  // layer2 only (else null)
{
  int wave = (int)((blockIdx.x * blockDim.x + threadIdx.x) >> 5);
  if (wave >= 2 * (HP / 16)) return;                   // uniform per-wave
  int lane  = (int)(threadIdx.x & 31);
  int ntile = wave % (HP / 16);
  int mtile = wave / (HP / 16);
  int l15   = lane & 15;
  int khalf = (lane >> 4) * 8;
  int arow  = mtile * 16 + l15;
  int brow  = ntile * 16 + l15;

  v8f z;
#pragma unroll
  for (int i = 0; i < 8; ++i) z[i] = 0.f;
  v8f acc[4] = {z, z, z, z};

  // ---- x @ Wih^T ----
  const bf16_t* ax = xin + (size_t)arow * ldx + khalf;
  for (int k = 0; k < Kx; k += 32) {
    v16bf a = load_frag(ax + k);
#pragma unroll
    for (int g = 0; g < 4; ++g) {
      const bf16_t* wp = wih + ((size_t)g * HP + brow) * (size_t)Kx + khalf + k;
      v16bf b = load_frag(wp);
      acc[g] = wmma_bf16(a, b, acc[g]);
    }
  }
  // ---- h @ Whh^T ----
  const bf16_t* ah = hprev + (size_t)arow * HP + khalf;
  for (int k = 0; k < HP; k += 32) {
    v16bf a = load_frag(ah + k);
#pragma unroll
    for (int g = 0; g < 4; ++g) {
      const bf16_t* wp = whh + ((size_t)g * HP + brow) * (size_t)HP + khalf + k;
      v16bf b = load_frag(wp);
      acc[g] = wmma_bf16(a, b, acc[g]);
    }
  }

  // ---- biases (n >= Hh lanes are padding; keep finite) ----
  int n = ntile * 16 + l15;
  float bi[4];
#pragma unroll
  for (int g = 0; g < 4; ++g)
    bi[g] = (n < Hh) ? (bih[g * Hh + n] + bhh[g * Hh + n]) : 0.f;

  // ---- cell update: C-fragment row r -> M = mtile*16 + khalf + r ----
#pragma unroll
  for (int r = 0; r < 8; ++r) {
    int m = mtile * 16 + khalf + r;
    float gi = acc[0][r] + bi[0];
    float gf = acc[1][r] + bi[1];
    float go = acc[2][r] + bi[2];
    float gg = acc[3][r] + bi[3];
    size_t idx = (size_t)m * HP + n;
    float c_old = c_state[idx];
    float c_new = sigm(gf) * c_old + sigm(gi) * tanhf(gg);
    float hval  = sigm(go) * tanhf(c_new);
    c_state[idx] = c_new;
    h_state[idx] = hval;
    hnew[idx]    = f2bf(hval);
    if (raw_bf)              raw_bf [(size_t)m * raw_bf_stride  + n] = f2bf(hval);
    if (raw_f32 && n < Hh)   raw_f32[(size_t)m * raw_f32_stride + n] = hval;
  }
}

// =====================================================================
// Block-cooperative bf16 GEMM  C = A @ W^T (+bias) with LDS-staged B.
// Block = 8 waves; block computes (8 M-tiles) x (NT N-tiles).  The NT*16
// B rows for the current 32-wide K slice are DMA'd global->LDS with
// global_load_async_to_lds_b128 (double buffered, ASYNCcnt + workgroup
// barrier), so B is read from L2 once per block instead of once per wave.
// A [M][lda] row-major, W [N][ldw] row-major (natural WMMA-B layout).
// =====================================================================
template <int NT>
__global__ void __launch_bounds__(256) gemm_lds_kernel(
    const bf16_t* __restrict__ A, int lda,
    const bf16_t* __restrict__ W, int ldw,
    const float*  __restrict__ bias,
    float*  __restrict__ Cf, long ldc,
    bf16_t* __restrict__ Cb, long ldcb,
    int K, int ngroups)
{
  __shared__ __align__(16) bf16_t smemB[2][NT * 16][32];   // 2 x NT KB

  int tid    = (int)threadIdx.x;
  int wave   = tid >> 5;
  int lane   = tid & 31;
  int mblock = (int)blockIdx.x / ngroups;
  int ng     = (int)blockIdx.x % ngroups;
  int mtile  = mblock * 8 + wave;
  int l15    = lane & 15;
  int khalf  = (lane >> 4) * 8;
  int nbase  = ng * (NT * 16);            // first B row of this group

  uint32_t lds0 = (uint32_t)(uintptr_t)(&smemB[0][0][0]);
  const uint32_t bufbytes = (uint32_t)(NT * 16 * 32 * 2);

  v8f z;
#pragma unroll
  for (int i = 0; i < 8; ++i) z[i] = 0.f;
  v8f acc[NT];
#pragma unroll
  for (int j = 0; j < NT; ++j) acc[j] = z;

  const bf16_t* ap = A + (size_t)(mtile * 16 + l15) * lda + khalf;
  const int nk = K / 32;
  const int nchunks = NT * 16 * 4;        // 16B chunks per 32-K B slice

  // ---- prologue: stage k-slice 0 into buffer 0 ----
  for (int c = tid; c < nchunks; c += 256) {
    int row = c >> 2, part = c & 3;
    async_g2l_b128(lds0 + (uint32_t)(row * 64 + part * 16),
                   W + (size_t)(nbase + row) * ldw + part * 8);
  }

  for (int ks = 0; ks < nk; ++ks) {
    wait_async0();          // my share of this slice has landed in LDS
    __syncthreads();        // everyone's share has landed
    if (ks + 1 < nk) {      // stage next slice into the other buffer
      uint32_t base = lds0 + (uint32_t)((ks + 1) & 1) * bufbytes;
      int kofs = (ks + 1) * 32;
      for (int c = tid; c < nchunks; c += 256) {
        int row = c >> 2, part = c & 3;
        async_g2l_b128(base + (uint32_t)(row * 64 + part * 16),
                       W + (size_t)(nbase + row) * ldw + kofs + part * 8);
      }
    }
    v16bf a = load_frag(ap + ks * 32);
    int buf = ks & 1;
#pragma unroll
    for (int j = 0; j < NT; ++j) {
      v16bf b = load_frag(&smemB[buf][j * 16 + l15][khalf]);   // ds_load_b128 x2
      acc[j] = wmma_bf16(a, b, acc[j]);
    }
    __syncthreads();        // all waves done reading buf before restage at ks+2
  }

#pragma unroll
  for (int j = 0; j < NT; ++j) {
    int nn = nbase + j * 16 + l15;
    float bv = bias ? bias[nn] : 0.f;
#pragma unroll
    for (int r = 0; r < 8; ++r) {
      int m = mtile * 16 + khalf + r;
      float v = acc[j][r] + bv;
      if (Cf) Cf[(size_t)m * ldc  + nn] = v;
      if (Cb) Cb[(size_t)m * ldcb + nn] = f2bf(v);
    }
  }
}

// =====================================================================
// Packing kernels (fp32 -> zero-padded bf16 tiles)
// =====================================================================
__global__ void pack_gate_kernel(const float* __restrict__ src,  // [4*Hh][Kin]
                                 bf16_t* __restrict__ dst,       // [4][HP][Kp]
                                 int Kin, int Kp)
{
  size_t total = (size_t)4 * HP * Kp;
  for (size_t i = blockIdx.x * (size_t)blockDim.x + threadIdx.x; i < total;
       i += (size_t)gridDim.x * blockDim.x) {
    int k = (int)(i % Kp);
    size_t t = i / Kp;
    int nn = (int)(t % HP);
    int g  = (int)(t / HP);
    float v = (nn < Hh && k < Kin) ? src[((size_t)g * Hh + nn) * Kin + k] : 0.f;
    dst[i] = f2bf(v);
  }
}

__global__ void pack_rows_kernel(const float* __restrict__ src,  // [rows][Kin]
                                 bf16_t* __restrict__ dst,       // [rows][Kp]
                                 int rows, int Kin, int Kp)
{
  size_t total = (size_t)rows * Kp;
  for (size_t i = blockIdx.x * (size_t)blockDim.x + threadIdx.x; i < total;
       i += (size_t)gridDim.x * blockDim.x) {
    int k = (int)(i % Kp);
    int r = (int)(i / Kp);
    float v = (k < Kin) ? src[(size_t)r * Kin + k] : 0.f;
    dst[i] = f2bf(v);
  }
}

__global__ void embed_pack_kernel(const int* __restrict__ x,      // [Bb][Ss]
                                  const float* __restrict__ embw, // [Vv][Ee]
                                  bf16_t* __restrict__ dst)       // [Ss][Bb][EP]
{
  size_t total = (size_t)Ss * Bb * EP;
  for (size_t i = blockIdx.x * (size_t)blockDim.x + threadIdx.x; i < total;
       i += (size_t)gridDim.x * blockDim.x) {
    int k = (int)(i % EP);
    size_t t = i / EP;
    int b = (int)(t % Bb);
    int s = (int)(t / Bb);
    int tok = x[(size_t)b * Ss + s];
    float v = (k < Ee) ? embw[(size_t)tok * Ee + k] : 0.f;
    dst[i] = f2bf(v);
  }
}

__global__ void copy_states_kernel(const float* __restrict__ hs,  // [3][Bb][HP]
                                   const float* __restrict__ cs,
                                   float* __restrict__ hout,      // [3][Bb][Hh]
                                   float* __restrict__ cout)
{
  size_t total = (size_t)3 * Bb * Hh;
  for (size_t i = blockIdx.x * (size_t)blockDim.x + threadIdx.x; i < total;
       i += (size_t)gridDim.x * blockDim.x) {
    int n = (int)(i % Hh);
    size_t t = i / Hh;
    int b = (int)(t % Bb);
    int l = (int)(t / Bb);
    size_t sidx = ((size_t)(l * Bb + b)) * HP + n;
    hout[i] = hs[sidx];
    cout[i] = cs[sidx];
  }
}

// =====================================================================
// Host orchestration
// =====================================================================
extern "C" void kernel_launch(void* const* d_in, const int* in_sizes, int n_in,
                              void* d_out, int out_size, void* d_ws, size_t ws_size,
                              hipStream_t stream)
{
  (void)in_sizes; (void)n_in; (void)out_size; (void)ws_size;

  const int*   x     = (const int*)  d_in[0];
  const float* embw  = (const float*)d_in[1];
  const float* wih0  = (const float*)d_in[2];
  const float* wih1  = (const float*)d_in[3];
  const float* wih2  = (const float*)d_in[4];
  const float* whh   = (const float*)d_in[5];
  const float* bih   = (const float*)d_in[6];
  const float* bhh   = (const float*)d_in[7];
  const float* projw = (const float*)d_in[8];
  const float* fcb   = (const float*)d_in[9];

  float* out_logits = (float*)d_out;                       // [Bb][Ss][Vv]
  float* out_h      = out_logits + (size_t)Bb * Ss * Vv;   // [3][Bb][Hh]
  float* out_c      = out_h + (size_t)3 * Bb * Hh;         // [3][Bb][Hh]
  float* out_drop   = out_c + (size_t)3 * Bb * Hh;         // [Bb][Ss][Ee]
  float* out_raw    = out_drop + (size_t)Bb * Ss * Ee;     // [Bb][Ss][Hh]

  // ---- workspace carve-out (256B aligned) ----
  char* ws = (char*)d_ws;
  size_t off = 0;
  auto carve = [&](size_t bytes) -> char* {
    char* p = ws + off;
    off += (bytes + 255) & ~(size_t)255;
    return p;
  };
  bf16_t* wih0g  = (bf16_t*)carve((size_t)4 * HP * EP * 2);
  bf16_t* wih1g  = (bf16_t*)carve((size_t)4 * HP * HP * 2);
  bf16_t* wih2g  = (bf16_t*)carve((size_t)4 * HP * HP * 2);
  bf16_t* whhg   = (bf16_t*)carve((size_t)3 * 4 * HP * HP * 2);
  bf16_t* projg  = (bf16_t*)carve((size_t)Ee * HP * 2);
  bf16_t* embwg  = (bf16_t*)carve((size_t)Vv * EP * 2);
  bf16_t* embx   = (bf16_t*)carve((size_t)Ss * Bb * EP * 2);
  bf16_t* rawbf  = (bf16_t*)carve((size_t)Bb * Ss * HP * 2);
  bf16_t* dropbf = (bf16_t*)carve((size_t)Bb * Ss * EP * 2);
  bf16_t* hbf0   = (bf16_t*)carve((size_t)3 * Bb * HP * 2);
  bf16_t* hbf1   = (bf16_t*)carve((size_t)3 * Bb * HP * 2);
  float*  hstate = (float*) carve((size_t)3 * Bb * HP * 4);
  float*  cstate = (float*) carve((size_t)3 * Bb * HP * 4);

  // ---- init (graph-capture-safe async memsets) ----
  hipMemsetAsync(hbf0,   0, (size_t)3 * Bb * HP * 2, stream);
  hipMemsetAsync(cstate, 0, (size_t)3 * Bb * HP * 4, stream);
  hipMemsetAsync(dropbf, 0, (size_t)Bb * Ss * EP * 2, stream);  // pad cols stay 0

  // ---- weight/activation packing to padded bf16 ----
  pack_gate_kernel<<<1024, 256, 0, stream>>>(wih0, wih0g, Ee, EP);
  pack_gate_kernel<<<2048, 256, 0, stream>>>(wih1, wih1g, Hh, HP);
  pack_gate_kernel<<<2048, 256, 0, stream>>>(wih2, wih2g, Hh, HP);
  for (int l = 0; l < 3; ++l)
    pack_gate_kernel<<<2048, 256, 0, stream>>>(
        whh + (size_t)l * 4 * Hh * Hh, whhg + (size_t)l * 4 * HP * HP, Hh, HP);
  pack_rows_kernel<<<1800, 256, 0, stream>>>(projw, projg, Ee, Hh, HP);
  pack_rows_kernel<<<4096, 256, 0, stream>>>(embw, embwg, Vv, Ee, EP);
  embed_pack_kernel<<<8192, 256, 0, stream>>>(x, embw, embx);

  // ---- recurrent sweep: 512 steps x 3 layers, h ping-ponged ----
  const size_t hslab = (size_t)Bb * HP;
  for (int t = 0; t < Ss; ++t) {
    bf16_t* hin  = (t & 1) ? hbf1 : hbf0;
    bf16_t* hout = (t & 1) ? hbf0 : hbf1;
    for (int l = 0; l < 3; ++l) {
      const bf16_t* xin;
      int ldx, Kx;
      const bf16_t* wihg;
      if (l == 0) { xin = embx + (size_t)t * Bb * EP; ldx = EP; Kx = EP; wihg = wih0g; }
      else        { xin = hout + (size_t)(l - 1) * hslab; ldx = HP; Kx = HP;
                    wihg = (l == 1) ? wih1g : wih2g; }
      float*  rawf = nullptr;
      bf16_t* rawb = nullptr;
      if (l == 2) { rawf = out_raw + (size_t)t * Hh; rawb = rawbf + (size_t)t * HP; }
      lstm_step_kernel<<<18, 256, 0, stream>>>(
          xin, ldx, Kx, wihg,
          hin + (size_t)l * hslab,
          whhg + (size_t)l * 4 * HP * HP,
          bih + (size_t)l * 4 * Hh, bhh + (size_t)l * 4 * Hh,
          cstate + (size_t)l * hslab,
          hstate + (size_t)l * hslab,
          hout + (size_t)l * hslab,
          rawf, (long)((size_t)Ss * Hh),
          rawb, (long)((size_t)Ss * HP));
    }
  }

  // ---- projection: dropped = raw @ proj_w^T (fp32 out + bf16 copy) ----
  // M blocks = 1024/8 = 128, N groups = 25/5 = 5 -> 640 blocks
  gemm_lds_kernel<5><<<640, 256, 0, stream>>>(
      rawbf, HP, projg, HP, nullptr,
      out_drop, (long)Ee, dropbf, (long)EP,
      HP, 5);

  // ---- logits = dropped @ emb_weight^T + fc_b ----
  // M blocks = 128, N groups = 625/5 = 125 -> 16000 blocks
  gemm_lds_kernel<5><<<16000, 256, 0, stream>>>(
      dropbf, EP, embwg, EP, fcb,
      out_logits, (long)Vv, (bf16_t*)nullptr, 0,
      EP, 125);

  // ---- final (h, c) states ----
  copy_states_kernel<<<432, 256, 0, stream>>>(hstate, cstate, out_h, out_c);
}
